// BMM_ABE_I8_7146825580667
// MI455X (gfx1250) — 8-bit matrix operand order repaired and checked by reading, compiled, not run on hardware
//
#include <hip/hip_runtime.h>
#include <stdint.h>

typedef int v8i __attribute__((ext_vector_type(8)));

constexpr int BB = 32;
constexpr int M  = 1024;
constexpr int K  = 1024;
constexpr int N  = 1024;

__global__ void pack_a_kernel(const int* __restrict__ src, uint8_t* __restrict__ dst) {
  const size_t n4 = (size_t)BB * M * K / 4;
  size_t i = (size_t)blockIdx.x * blockDim.x + threadIdx.x;
  if (i >= n4) return;
  int4 v = ((const int4*)src)[i];
  uint32_t p = (uint32_t)(v.x & 0xff)
             | ((uint32_t)(v.y & 0xff) << 8)
             | ((uint32_t)(v.z & 0xff) << 16)
             | ((uint32_t)(v.w & 0xff) << 24);
  ((uint32_t*)dst)[i] = p;
}

__global__ void pack_bt_kernel(const int* __restrict__ src, int8_t* __restrict__ dst) {
  __shared__ int8_t tile[32][33];
  const int b  = blockIdx.z;
  const int k0 = blockIdx.y * 32;
  const int n0 = blockIdx.x * 32;
  #pragma unroll
  for (int r = 0; r < 32; r += 8) {
    int k = k0 + threadIdx.y + r;
    int n = n0 + threadIdx.x;
    tile[threadIdx.y + r][threadIdx.x] = (int8_t)src[((size_t)b * K + k) * N + n];
  }
  __syncthreads();
  #pragma unroll
  for (int r = 0; r < 32; r += 8) {
    int n = n0 + threadIdx.y + r;
    int k = k0 + threadIdx.x;
    dst[((size_t)b * N + n) * K + k] = tile[threadIdx.x][threadIdx.y + r];
  }
}

__device__ __forceinline__ v8i load_a_frag(const int8_t* p) {
  int2 q0 = *(const int2*)(p + 0);
  int2 q1 = *(const int2*)(p + 16);
  int2 q2 = *(const int2*)(p + 32);
  int2 q3 = *(const int2*)(p + 48);
  v8i r;
  r[0] = q0.x; r[1] = q0.y;
  r[2] = q1.x; r[3] = q1.y;
  r[4] = q2.x; r[5] = q2.y;
  r[6] = q3.x; r[7] = q3.y;
  return r;
}

__device__ __forceinline__ v8i load_b_frag(const int8_t* p) {
  int2 q0 = *(const int2*)(p + 0);
  int2 q1 = *(const int2*)(p + 16);
  int2 q2 = *(const int2*)(p + 32);
  int2 q3 = *(const int2*)(p + 48);
  v8i r;
  r[0] = q0.x; r[1] = q0.y; r[2] = q1.x; r[3] = q1.y;
  r[4] = q2.x; r[5] = q2.y; r[6] = q3.x; r[7] = q3.y;
  return r;
}

__global__ void __launch_bounds__(128)
bmm_i8_wmma_kernel(const uint8_t* __restrict__ a8,
                   const int8_t*  __restrict__ bt8,
                   const float*   __restrict__ alpha_p,
                   float*         __restrict__ out) {
  const int lane = threadIdx.x & 31;
  const int wave = threadIdx.x >> 5;
  const int hi   = lane >> 4;
  const int ln   = lane & 15;

  const int b     = blockIdx.z;
  const int nTile = blockIdx.x * 32;
  const int mTile = blockIdx.y * 128 + wave * 32;

  const int8_t* Ab = (const int8_t*)a8 + (size_t)b * M * K;
  const int8_t* Bb = bt8 + (size_t)b * N * K;

  const int8_t* aRow0 = Ab + (size_t)(mTile + ln) * K + hi * 8;
  const int8_t* aRow1 = Ab + (size_t)(mTile + 16 + ln) * K + hi * 8;
  const int8_t* bCol0 = Bb + (size_t)(nTile + ln) * K + hi * 8;
  const int8_t* bCol1 = Bb + (size_t)(nTile + 16 + ln) * K + hi * 8;

  v8i acc00 = {}, acc01 = {}, acc10 = {}, acc11 = {};

  for (int k0 = 0; k0 < K; k0 += 64) {
    v8i a0 = load_a_frag(aRow0 + k0);
    v8i a1 = load_a_frag(aRow1 + k0);
    v8i b0 = load_b_frag(bCol0 + k0);
    v8i b1 = load_b_frag(bCol1 + k0);

    if (k0 + 64 < K) {
      __builtin_prefetch(aRow0 + k0 + 64, 0, 1);
      __builtin_prefetch(bCol0 + k0 + 64, 0, 1);
    }

    acc00 = __builtin_amdgcn_wmma_i32_16x16x64_iu8(false, a0, false, b0, acc00, false, false);
    acc01 = __builtin_amdgcn_wmma_i32_16x16x64_iu8(false, a0, false, b1, acc01, false, false);
    acc10 = __builtin_amdgcn_wmma_i32_16x16x64_iu8(false, a1, false, b0, acc10, false, false);
    acc11 = __builtin_amdgcn_wmma_i32_16x16x64_iu8(false, a1, false, b1, acc11, false, false);
  }

  const float alpha = alpha_p[0];
  float* outB = out + (size_t)b * M * N;
  #pragma unroll
  for (int v = 0; v < 8; ++v) {
    const int m0 = mTile + v + hi * 8;
    const int n  = nTile + ln;
    outB[(size_t)m0 * N + n]             = (float)acc00[v] * alpha;
    outB[(size_t)m0 * N + n + 16]        = (float)acc01[v] * alpha;
    outB[(size_t)(m0 + 16) * N + n]      = (float)acc10[v] * alpha;
    outB[(size_t)(m0 + 16) * N + n + 16] = (float)acc11[v] * alpha;
  }
}

extern "C" void kernel_launch(void* const* d_in, const int* in_sizes, int n_in,
                              void* d_out, int out_size, void* d_ws, size_t ws_size,
                              hipStream_t stream) {
  const int*   a_i32 = (const int*)d_in[0];
  const int*   b_i32 = (const int*)d_in[1];
  const float* alpha = (const float*)d_in[2];
  float*       out   = (float*)d_out;

  uint8_t* a8  = (uint8_t*)d_ws;
  int8_t*  bt8 = (int8_t*)d_ws + (size_t)BB * M * K;

  {
    size_t n4 = (size_t)BB * M * K / 4;
    int threads = 256;
    int blocks = (int)((n4 + threads - 1) / threads);
    pack_a_kernel<<<blocks, threads, 0, stream>>>(a_i32, a8);
  }
  {
    dim3 grid(N / 32, K / 32, BB);
    dim3 block(32, 8);
    pack_bt_kernel<<<grid, block, 0, stream>>>(b_i32, bt8);
  }
  {
    dim3 grid(N / 32, M / 128, BB);
    dim3 block(128);
    bmm_i8_wmma_kernel<<<grid, block, 0, stream>>>(a8, bt8, alpha, out);
  }
}
